// FCOSModule_78460462564098
// MI455X (gfx1250) — compile-verified, matrix operand order checked by reading, not run on hardware
//
#include <hip/hip_runtime.h>
#include <hip/hip_bf16.h>

typedef __bf16 bf16;
typedef bf16  v16bf __attribute__((ext_vector_type(16)));
typedef bf16  v8bf  __attribute__((ext_vector_type(8)));
typedef float v8f   __attribute__((ext_vector_type(8)));

#define KTOT   2304   // 9 taps * 256 in-channels
#define CHUNK  128    // K elems staged in LDS per iteration
#define NCHUNK 18     // 2304 / 128
#define LDSTRIDE 136  // bf16 per LDS position row (128 + 8 pad; 16B aligned rows, rotates banks)

// ---------------------------------------------------------------------------
// Pack conv weights (CoutReal x 256 x 3 x 3, f32) -> (CoutPad x 2304, bf16)
// k ordering: k = (ky*3+kx)*256 + ci ; rows beyond CoutReal are zero.
// ---------------------------------------------------------------------------
__global__ __launch_bounds__(256) void fcos_pack_w(const float* __restrict__ Wsrc,
                                                   bf16* __restrict__ Wpk,
                                                   int CoutReal, int CoutPad) {
  int idx = blockIdx.x * 256 + threadIdx.x;
  int total = CoutPad * KTOT;
  if (idx >= total) return;
  int m = idx / KTOT;
  int k = idx - m * KTOT;
  int r = k >> 8;      // tap 0..8
  int ci = k & 255;
  float v = 0.f;
  if (m < CoutReal) v = Wsrc[((long)m * 256 + ci) * 9 + r];
  Wpk[idx] = (bf16)v;
}

// ---------------------------------------------------------------------------
// Implicit-GEMM 3x3 conv (pad=1) via bf16 WMMA.
// Block = 128 threads (4 waves).
//   JT   : Cout tiles per wave (4 -> block covers 256 Cout, all statically valid)
//   CPAD : padded Cout (multiple of 16)
//   CREAL: real Cout (store/bias guard; folds away when == CPAD)
// Wave w owns tiles (blockIdx.y*4*JT + w + 4*j), j = 0..JT-1.
// blockIdx.x = n*tilesP + tileP (16 positions per tile).
// ---------------------------------------------------------------------------
template <int JT, int CPAD, int CREAL>
__global__ __launch_bounds__(128) void fcos_conv_gemm(
    const float* __restrict__ X, const bf16* __restrict__ Wpk,
    const float* __restrict__ bias, float* __restrict__ Out,
    int H, int W, long out_n_stride, long out_c_stride, long out_col_off) {
  constexpr bool FULL = (JT == 4);  // every tile statically valid (CPAD == 256)
  static_assert(FULL || JT == 1, "JT must be 4 (full) or 1 (guarded)");

  __shared__ bf16 lds_b[16 * LDSTRIDE];

  const int HW = H * W;
  const int tilesP = (HW + 15) >> 4;
  const int n_img = blockIdx.x / tilesP;
  const int tp = blockIdx.x - n_img * tilesP;
  const int p0 = tp * 16;

  const int t = threadIdx.x;
  const int wave = t >> 5;
  const int lane = t & 31;
  const int nlo = lane & 15;   // N column (positions) / M row (weights) per WMMA layout
  const int half = lane >> 4;  // K-half selector

  int m0s[JT];
#pragma unroll
  for (int j = 0; j < JT; ++j) m0s[j] = (blockIdx.y * 4 * JT + wave + 4 * j) * 16;
  const bool act = FULL || (m0s[0] < CPAD);  // wave-level guard (JT==1 path)

  // LDS-fill assignment: thread fills pos = t/8, 16 consecutive k's.
  // k-runs are 16-aligned and tap boundaries sit at multiples of 256, so the
  // tap (and its spatial-bounds predicate) is uniform across the run.
  const int fpos = t >> 3;
  const int fk0 = (t & 7) * 16;
  const int p_f = p0 + fpos;
  const bool pvalid = (p_f < HW);
  int py = 0, px = 0;
  if (pvalid) { py = p_f / W; px = p_f - py * W; }
  const float* Xn = X + (long)n_img * 256 * HW;

  v8f acc[JT];
#pragma unroll
  for (int j = 0; j < JT; ++j) acc[j] = (v8f){};

  for (int c = 0; c < NCHUNK; ++c) {
    __syncthreads();
    // ---- stage B chunk (im2col gather, f32 -> bf16) into LDS [pos][k] ----
    {
      const int kg = c * CHUNK + fk0;
      const int r = kg >> 8;      // tap, constant for the whole 16-run
      const int ci0 = kg & 255;   // first in-channel of the run
      const int iy = py + (r / 3) - 1;
      const int ix = px + (r % 3) - 1;
      const bool ok = pvalid && iy >= 0 && iy < H && ix >= 0 && ix < W;
      // clamped base address is always in-bounds -> branch-free loads + select
      const float* srcp = Xn + (long)ci0 * HW + (ok ? (iy * W + ix) : 0);
      v8bf pk0, pk1;
#pragma unroll
      for (int i = 0; i < 8; ++i) {
        float v = srcp[(long)i * HW];
        pk0[i] = (bf16)(ok ? v : 0.f);
      }
#pragma unroll
      for (int i = 0; i < 8; ++i) {
        float v = srcp[(long)(i + 8) * HW];
        pk1[i] = (bf16)(ok ? v : 0.f);
      }
      *(v8bf*)&lds_b[fpos * LDSTRIDE + fk0] = pk0;
      *(v8bf*)&lds_b[fpos * LDSTRIDE + fk0 + 8] = pk1;
    }
    __syncthreads();

    if (act) {
#pragma unroll
      for (int s2 = 0; s2 < 4; ++s2) {
        const int kb = s2 * 32 + half * 8;
        // B fragment: lane holds column nlo; shared across all JT Cout tiles
        v8bf blo = *(const v8bf*)(&lds_b[nlo * LDSTRIDE + kb]);
        v8bf bhi = *(const v8bf*)(&lds_b[nlo * LDSTRIDE + kb + 16]);
        v16bf b = __builtin_shufflevector(blo, bhi, 0,1,2,3,4,5,6,7,8,9,10,11,12,13,14,15);
#pragma unroll
        for (int j = 0; j < JT; ++j) {
          const bf16* arow = Wpk + (long)(m0s[j] + nlo) * KTOT + c * CHUNK;
          if (s2 == 0 && c + 1 < NCHUNK) __builtin_prefetch(arow + CHUNK, 0, 3);
          v8bf alo = *(const v8bf*)(arow + kb);
          v8bf ahi = *(const v8bf*)(arow + kb + 16);
          v16bf a = __builtin_shufflevector(alo, ahi, 0,1,2,3,4,5,6,7,8,9,10,11,12,13,14,15);
          acc[j] = __builtin_amdgcn_wmma_f32_16x16x32_bf16(
              /*neg_a=*/false, a, /*neg_b=*/false, b,
              /*c_mod=*/(short)0, acc[j], /*reuse_a=*/false, /*reuse_b=*/false);
        }
      }
    }
  }

  // ---- store: acc[j][r] -> (M = m0 + r + 8*half, N = nlo) ----
  const int p = p0 + nlo;
  if (act && p < HW) {
#pragma unroll
    for (int j = 0; j < JT; ++j) {
#pragma unroll
      for (int r = 0; r < 8; ++r) {
        const int cout = m0s[j] + r + half * 8;
        if (CPAD == CREAL || cout < CREAL) {
          Out[(long)n_img * out_n_stride + (long)cout * out_c_stride + out_col_off + p] =
              acc[j][r] + bias[cout];
        }
      }
    }
  }
}

// ---------------------------------------------------------------------------
// GroupNorm stats: one block per (n, group). group = 8 contiguous channels.
// stats[b] = {mean, rstd}
// ---------------------------------------------------------------------------
__global__ __launch_bounds__(256) void fcos_gn_stats(const float* __restrict__ T,
                                                     float* __restrict__ stats, int HW) {
  __shared__ float s1[256];
  __shared__ float s2[256];
  const int b = blockIdx.x;
  const int n = b >> 5;
  const int g = b & 31;
  const float* base = T + ((long)n * 256 + (long)g * 8) * HW;
  const long cnt = 8L * HW;
  float a = 0.f, q = 0.f;
  for (long i = threadIdx.x; i < cnt; i += 256) {
    float v = base[i];
    a += v; q += v * v;
  }
  s1[threadIdx.x] = a; s2[threadIdx.x] = q;
  __syncthreads();
  for (int off = 128; off > 0; off >>= 1) {
    if ((int)threadIdx.x < off) {
      s1[threadIdx.x] += s1[threadIdx.x + off];
      s2[threadIdx.x] += s2[threadIdx.x + off];
    }
    __syncthreads();
  }
  if (threadIdx.x == 0) {
    float m = s1[0] / (float)cnt;
    float var = s2[0] / (float)cnt - m * m;
    stats[b * 2 + 0] = m;
    stats[b * 2 + 1] = rsqrtf(var + 1e-5f);
  }
}

// ---------------------------------------------------------------------------
// GroupNorm apply + ReLU (elementwise over N*256*HW)
// ---------------------------------------------------------------------------
__global__ __launch_bounds__(256) void fcos_gn_relu(const float* __restrict__ T,
                                                    const float* __restrict__ stats,
                                                    const float* __restrict__ gw,
                                                    const float* __restrict__ gb,
                                                    float* __restrict__ Out, int HW) {
  int idx = blockIdx.x * 256 + threadIdx.x;
  int total = 2 * 256 * HW;
  if (idx >= total) return;
  int c = (idx / HW) & 255;
  int n = idx / (256 * HW);
  int g = c >> 3;
  int sb = (n * 32 + g) * 2;
  float m = stats[sb];
  float rs = stats[sb + 1];
  float v = (T[idx] - m) * rs * gw[c] + gb[c];
  Out[idx] = v > 0.f ? v : 0.f;
}

// ---------------------------------------------------------------------------
// Locations + anchor sizes for one level
// ---------------------------------------------------------------------------
__global__ __launch_bounds__(256) void fcos_loc_anchor(float* __restrict__ loc,
                                                       float* __restrict__ anch,
                                                       int H, int W, int s, int off) {
  int p = blockIdx.x * 256 + threadIdx.x;
  if (p >= H * W) return;
  int py = p / W;
  int px = p - py * W;
  loc[(long)(off + p) * 2 + 0] = (float)(px * s + s / 2);
  loc[(long)(off + p) * 2 + 1] = (float)(py * s + s / 2);
  anch[off + p] = (float)(s * 4);
}

// ---------------------------------------------------------------------------
extern "C" void kernel_launch(void* const* d_in, const int* in_sizes, int n_in,
                              void* d_out, int out_size, void* d_ws, size_t ws_size,
                              hipStream_t stream) {
  (void)in_sizes; (void)n_in; (void)out_size; (void)ws_size;

  const float* feat[5];
  for (int i = 0; i < 5; ++i) feat[i] = (const float*)d_in[i];
  const float* cls_w    = (const float*)d_in[5];
  const float* cls_b    = (const float*)d_in[6];
  const float* cls_gn_w = (const float*)d_in[7];
  const float* cls_gn_b = (const float*)d_in[8];
  const float* bbox_w   = (const float*)d_in[9];
  const float* bbox_b   = (const float*)d_in[10];
  const float* bbox_gn_w= (const float*)d_in[11];
  const float* bbox_gn_b= (const float*)d_in[12];
  const float* logits_w = (const float*)d_in[13];
  const float* logits_b = (const float*)d_in[14];
  const float* pred_w   = (const float*)d_in[15];
  const float* pred_b   = (const float*)d_in[16];

  const int TOT = 17064;
  float* out = (float*)d_out;
  float* logits_out = out;                       // [2, 80, 17064]
  float* bbox_out   = out + 2L * 80 * TOT;       // [2, 4, 17064]
  float* loc_out    = bbox_out + 2L * 4 * TOT;   // [17064, 2]
  float* anch_out   = loc_out + 2L * TOT;        // [17064, 1]

  // ---- workspace carve-up ----
  char* ws = (char*)d_ws;
  size_t off_b = 0;
  auto walloc = [&](size_t bytes) -> void* {
    void* p = ws + off_b;
    off_b = (off_b + bytes + 255) & ~(size_t)255;
    return p;
  };
  bf16* wpk_cls  = (bf16*)walloc(4UL * 256 * KTOT * sizeof(bf16));
  bf16* wpk_bbox = (bf16*)walloc(4UL * 256 * KTOT * sizeof(bf16));
  bf16* wpk_log  = (bf16*)walloc(80UL * KTOT * sizeof(bf16));
  bf16* wpk_prd  = (bf16*)walloc(16UL * KTOT * sizeof(bf16));
  float* stats   = (float*)walloc(64 * 2 * sizeof(float));
  float* bufT    = (float*)walloc(2UL * 256 * 12800 * sizeof(float));
  float* bufA    = (float*)walloc(2UL * 256 * 12800 * sizeof(float));

  // ---- pack weights to bf16 GEMM layout ----
  {
    int g256 = (256 * KTOT + 255) / 256;
    for (int i = 0; i < 4; ++i) {
      fcos_pack_w<<<g256, 256, 0, stream>>>(cls_w  + (long)i * 256 * 256 * 9,
                                            wpk_cls  + (long)i * 256 * KTOT, 256, 256);
      fcos_pack_w<<<g256, 256, 0, stream>>>(bbox_w + (long)i * 256 * 256 * 9,
                                            wpk_bbox + (long)i * 256 * KTOT, 256, 256);
    }
    fcos_pack_w<<<(80 * KTOT + 255) / 256, 256, 0, stream>>>(logits_w, wpk_log, 80, 80);
    fcos_pack_w<<<(16 * KTOT + 255) / 256, 256, 0, stream>>>(pred_w,   wpk_prd,  4, 16);
  }

  static const int Hs[5] = {100, 50, 25, 13, 7};
  static const int Ws5[5] = {128, 64, 32, 16, 8};
  static const int Ss[5] = {8, 16, 32, 64, 128};
  int offs[5];
  {
    int a = 0;
    for (int l = 0; l < 5; ++l) { offs[l] = a; a += Hs[l] * Ws5[l]; }
  }

  for (int l = 0; l < 5; ++l)
    fcos_loc_anchor<<<(Hs[l] * Ws5[l] + 255) / 256, 256, 0, stream>>>(
        loc_out, anch_out, Hs[l], Ws5[l], Ss[l], offs[l]);

  for (int l = 0; l < 5; ++l) {
    const int H = Hs[l], W = Ws5[l], HW = H * W;
    const int tilesP = (HW + 15) / 16;
    dim3 blk(128);

    // ---- cls tower -> logits ----
    const float* src = feat[l];
    for (int i = 0; i < 4; ++i) {
      fcos_conv_gemm<4, 256, 256><<<dim3(2 * tilesP), blk, 0, stream>>>(
          src, wpk_cls + (long)i * 256 * KTOT, cls_b + i * 256, bufT,
          H, W, 256L * HW, (long)HW, 0L);
      fcos_gn_stats<<<64, 256, 0, stream>>>(bufT, stats, HW);
      fcos_gn_relu<<<(2 * 256 * HW + 255) / 256, 256, 0, stream>>>(
          bufT, stats, cls_gn_w + i * 256, cls_gn_b + i * 256, bufA, HW);
      src = bufA;
    }
    fcos_conv_gemm<1, 80, 80><<<dim3(2 * tilesP, 2), blk, 0, stream>>>(
        src, wpk_log, logits_b, logits_out,
        H, W, 80L * TOT, (long)TOT, (long)offs[l]);

    // ---- bbox tower -> pred ----
    src = feat[l];
    for (int i = 0; i < 4; ++i) {
      fcos_conv_gemm<4, 256, 256><<<dim3(2 * tilesP), blk, 0, stream>>>(
          src, wpk_bbox + (long)i * 256 * KTOT, bbox_b + i * 256, bufT,
          H, W, 256L * HW, (long)HW, 0L);
      fcos_gn_stats<<<64, 256, 0, stream>>>(bufT, stats, HW);
      fcos_gn_relu<<<(2 * 256 * HW + 255) / 256, 256, 0, stream>>>(
          bufT, stats, bbox_gn_w + i * 256, bbox_gn_b + i * 256, bufA, HW);
      src = bufA;
    }
    fcos_conv_gemm<1, 16, 4><<<dim3(2 * tilesP), blk, 0, stream>>>(
        src, wpk_prd, pred_b, bbox_out,
        H, W, 4L * TOT, (long)TOT, (long)offs[l]);
  }
}
